// TTMLinear_22351009808655
// MI455X (gfx1250) — matrix operand order repaired and checked by reading, not compiled, not run on hardware
//
#include <hip/hip_runtime.h>

typedef __bf16 bf16_t;
typedef __attribute__((ext_vector_type(16))) __bf16 v16bf;
typedef __attribute__((ext_vector_type(8)))  float  v8f;
typedef __attribute__((ext_vector_type(4)))  unsigned int u32x4;

#if defined(__AMDGCN__) && __has_builtin(__builtin_amdgcn_tensor_load_to_lds) && \
    __has_builtin(__builtin_amdgcn_s_wait_tensorcnt)
#define USE_TDM 1
#else
#define USE_TDM 0
#endif

// ---------------------------------------------------------------------------
// Step 1: TT core contraction chain (tiny).
// Prefix: cores 0..5  -> P[i6][j6][r], (64,64,16). Core0 init is a flat copy.
// Suffix: cores 6..11 -> S[r][i4][j6], (16,16,64). Core11 init is a flat copy.
// ---------------------------------------------------------------------------

// in: (IJ, IJ, 16)  G: (16,2,2,16)  out: (2IJ, 2IJ, 16)
__global__ void prefix_step_kernel(float* __restrict__ out, const float* __restrict__ in,
                                   const float* __restrict__ G, int IJ) {
  const int total = 4 * IJ * IJ * 16;
  const int e = blockIdx.x * blockDim.x + threadIdx.x;
  if (e >= total) return;
  const int r2 = e & 15;
  int t = e >> 4;
  const int J2 = 2 * IJ;
  const int j2 = t % J2;
  const int i2 = t / J2;
  const int m = i2 & 1, i = i2 >> 1;
  const int n = j2 & 1, j = j2 >> 1;
  const float* ip = in + (i * IJ + j) * 16;
  const float* gp = G + m * 32 + n * 16 + r2;   // G[r][m][n][r2], stride 64 in r
  float s = 0.f;
#pragma unroll
  for (int r = 0; r < 16; ++r) s += ip[r] * gp[r * 64];
  out[e] = s;
}

// in: (16, Iin, Jin)  G: (16, m, 2, 16)  out: (16, m*Iin, 2*Jin)
__global__ void suffix_step_kernel(float* __restrict__ out, const float* __restrict__ in,
                                   const float* __restrict__ G, int m, int Iin, int Jin) {
  const int Iout = m * Iin, Jout = 2 * Jin;
  const int total = 16 * Iout * Jout;
  const int e = blockIdx.x * blockDim.x + threadIdx.x;
  if (e >= total) return;
  const int j2 = e % Jout;
  int t = e / Jout;
  const int i2 = t % Iout;
  const int r  = t / Iout;
  const int ni = j2 / Jin, j = j2 % Jin;
  const int mi = i2 / Iin, i = i2 % Iin;
  const float* gp = G + ((r * m + mi) * 2 + ni) * 16;   // sum over r' (16)
  const float* ip = in + i * Jin + j;                   // stride Iin*Jin in r'
  const int stride = Iin * Jin;
  float s = 0.f;
#pragma unroll
  for (int rp = 0; rp < 16; ++rp) s += gp[rp] * ip[rp * stride];
  out[e] = s;
}

// ---------------------------------------------------------------------------
// Step 2: materialize W (1024x4096) = P x_r S, transposed to Wt[n][k] and
// split into bf16 hi/lo planes. One block per i_hi (64 blocks).
// ---------------------------------------------------------------------------
__global__ __launch_bounds__(256) void build_w_kernel(const float* __restrict__ P,
                                                      const float* __restrict__ S,
                                                      bf16_t* __restrict__ Wth,
                                                      bf16_t* __restrict__ Wtl) {
  __shared__ float sS[16 * 16 * 64];  // 64 KB
  __shared__ float sP[64 * 16];       // 4 KB: P[i_hi][j_hi][r]
  const int ihi = blockIdx.x;         // 0..63
  for (int e = threadIdx.x; e < 16 * 16 * 64; e += 256) sS[e] = S[e];
  for (int e = threadIdx.x; e < 64 * 16; e += 256) sP[e] = P[ihi * 1024 + e];
  __syncthreads();
  for (int e = threadIdx.x; e < 65536; e += 256) {
    const int ilo = e & 15;
    const int j = e >> 4;            // 0..4095
    const int jhi = j >> 6, jlo = j & 63;
    float s = 0.f;
#pragma unroll
    for (int r = 0; r < 16; ++r)
      s += sP[jhi * 16 + r] * sS[(r * 16 + ilo) * 64 + jlo];
    const int i = ihi * 16 + ilo;    // k index 0..1023
    const bf16_t h = (bf16_t)s;
    const bf16_t l = (bf16_t)(s - (float)h);
    Wth[(size_t)j * 1024 + i] = h;
    Wtl[(size_t)j * 1024 + i] = l;
  }
}

// ---------------------------------------------------------------------------
// Step 3: pad x 1000->1024 and split into bf16 hi/lo planes (4096 x 1024).
// ---------------------------------------------------------------------------
__global__ __launch_bounds__(256) void split_x_kernel(const float* __restrict__ x,
                                                      bf16_t* __restrict__ Xh,
                                                      bf16_t* __restrict__ Xl) {
  const size_t idx = (size_t)blockIdx.x * 256 + threadIdx.x;  // over 4096*1024
  const int k = (int)(idx & 1023);
  const size_t m = idx >> 10;
  const float v = (k < 1000) ? x[m * 1000 + k] : 0.f;
  const bf16_t h = (bf16_t)v;
  const bf16_t l = (bf16_t)(v - (float)h);
  Xh[idx] = h;
  Xl[idx] = l;
}

// ---------------------------------------------------------------------------
// Step 4: bf16x3 WMMA GEMM with TDM double-buffered LDS staging.
// Block tile 128x128, 8 waves (2x4), wave tile 64x32 (4x2 accum tiles),
// K-step 32, LDS pitch 40 elems (80B, via TDM pad) -> conflict-free b128.
// ---------------------------------------------------------------------------
#define GPITCH 40

__device__ inline v16bf frag_a(const bf16_t* rowbase, int sel) {
  // lane<16: K 0..7 and 16..23 ; lane>=16: K 8..15 and 24..31
  union { v16bf v; u32x4 q[2]; } f;
  f.q[0] = *(const u32x4*)(rowbase + sel * 8);
  f.q[1] = *(const u32x4*)(rowbase + 16 + sel * 8);
  return f.v;
}

__device__ inline v16bf frag_b(const bf16_t* rowbase, int sel) {
  // lane<16: K 0..7 and 16..23 ; lane>=16: K 8..15 and 24..31
  union { v16bf v; u32x4 q[2]; } f;
  f.q[0] = *(const u32x4*)(rowbase + sel * 8);
  f.q[1] = *(const u32x4*)(rowbase + 16 + sel * 8);
  return f.v;
}

#if USE_TDM
typedef __attribute__((ext_vector_type(4))) unsigned int tdm_g0_t;
typedef __attribute__((ext_vector_type(8))) int tdm_g1_t;
typedef __attribute__((ext_vector_type(4))) int tdm_g2_t;
typedef __attribute__((ext_vector_type(8))) int tdm_g4_t;

// One 128-row x 32-elem bf16 tile, global row stride 1024 elems, into LDS
// with 16B padding after every 64B row (=> 80B pitch == GPITCH elems).
__device__ inline void tdm_tile(const bf16_t* g, const bf16_t* lds) {
  const unsigned long long ga = (unsigned long long)g;
  const unsigned loff =
      (unsigned)(unsigned long long)(__attribute__((address_space(3))) const bf16_t*)lds;
  // D# group0: count=1 | lds_addr | global_addr[56:0] | type=2
  tdm_g0_t g0 = { 1u, loff, (unsigned)ga,
                  (unsigned)((ga >> 32) & 0x01FFFFFFull) | 0x80000000u };
  // D# group1: data_size=2B, pad_enable, pad_interval=16DW(64B), pad_amount=4DW(16B)
  //            tensor_dim0=tensor_dim1=2^30 (no OOB), tile 32x128, dim0_stride=1024
  tdm_g1_t g1 = { (int)0x06D10000u,  // wgmask=0|ds=1|pad=1|pint=3|pamt=3
                  (int)0x00000000u,  // atomic_barrier=0 | dim0[15:0]=0
                  (int)0x00004000u,  // dim0[31:16]=0x4000 | dim1[15:0]=0
                  (int)0x00204000u,  // dim1[31:16]=0x4000 | tile_dim0=32
                  128,               // tile_dim1=128 | tile_dim2=0
                  1024,              // dim0_stride[31:0]
                  0, 0 };            // dim0_stride[47:32]=0 | dim1_stride=0
  tdm_g2_t gz = { 0, 0, 0, 0 };
  tdm_g4_t gz8 = { 0, 0, 0, 0, 0, 0, 0, 0 };
  __builtin_amdgcn_tensor_load_to_lds(g0, g1, gz, gz, gz8, 0);
}
#endif

__global__ __launch_bounds__(256) void gemm_kernel(const bf16_t* __restrict__ Xh,
                                                   const bf16_t* __restrict__ Xl,
                                                   const bf16_t* __restrict__ Wth,
                                                   const bf16_t* __restrict__ Wtl,
                                                   float* __restrict__ Y) {
  // [buf][plane: Ah,Al,Bh,Bl][tile] : 2*4*10240B = 80KB
  __shared__ alignas(16) bf16_t smem[2][4][128 * GPITCH];

  const int tid  = threadIdx.x;
  const int lane = tid & 31;
  const int wave = tid >> 5;
  const int sel  = lane >> 4;   // half-wave select
  const int l16  = lane & 15;
  const int bm = blockIdx.y * 128;
  const int bn = blockIdx.x * 128;
  const int wm = (wave >> 2) * 64;   // wave row offset in block tile
  const int wn = (wave & 3) * 32;    // wave col offset in block tile

  const v8f vzero = {0.f, 0.f, 0.f, 0.f, 0.f, 0.f, 0.f, 0.f};
  v8f acc[4][2];
#pragma unroll
  for (int mt = 0; mt < 4; ++mt)
#pragma unroll
    for (int nt = 0; nt < 2; ++nt) acc[mt][nt] = vzero;

#if USE_TDM
  const bf16_t* gplane[4] = { Xh  + (size_t)bm * 1024, Xl  + (size_t)bm * 1024,
                              Wth + (size_t)bn * 1024, Wtl + (size_t)bn * 1024 };
  if (wave == 0) {
#pragma unroll
    for (int pl = 0; pl < 4; ++pl) tdm_tile(gplane[pl], &smem[0][pl][0]);
  }
#endif

  int buf = 0;
  for (int kt = 0; kt < 1024; kt += 32, buf ^= 1) {
#if USE_TDM
    // Wave 0: prefetch next tile via TDM into the other buffer, then ensure
    // the current buffer's 4 DMAs (in-order on TENSORcnt) have landed.
    if (wave == 0) {
      if (kt + 32 < 1024) {
#pragma unroll
        for (int pl = 0; pl < 4; ++pl)
          tdm_tile(gplane[pl] + kt + 32, &smem[buf ^ 1][pl][0]);
        __builtin_amdgcn_s_wait_tensorcnt(4);
      } else {
        __builtin_amdgcn_s_wait_tensorcnt(0);
      }
    }
#else
    // Fallback: cooperative synchronous fill of smem[buf].
#pragma unroll
    for (int p = 0; p < 2; ++p) {
      const int chunk = tid + p * 256;      // 0..511
      const int r = chunk >> 2;             // 0..127
      const int c = (chunk & 3) * 8;        // 0,8,16,24 elems
      const size_t ga = (size_t)(bm + r) * 1024 + kt + c;
      const size_t gb = (size_t)(bn + r) * 1024 + kt + c;
      *(u32x4*)&smem[buf][0][r * GPITCH + c] = *(const u32x4*)&Xh[ga];
      *(u32x4*)&smem[buf][1][r * GPITCH + c] = *(const u32x4*)&Xl[ga];
      *(u32x4*)&smem[buf][2][r * GPITCH + c] = *(const u32x4*)&Wth[gb];
      *(u32x4*)&smem[buf][3][r * GPITCH + c] = *(const u32x4*)&Wtl[gb];
    }
#endif
    __syncthreads();

    // ---- fragments ----
    v16bf ah[4], al[4], bh[2], bl[2];
#pragma unroll
    for (int mt = 0; mt < 4; ++mt) {
      const int row = wm + mt * 16 + l16;
      ah[mt] = frag_a(&smem[buf][0][row * GPITCH], sel);
      al[mt] = frag_a(&smem[buf][1][row * GPITCH], sel);
    }
#pragma unroll
    for (int nt = 0; nt < 2; ++nt) {
      const int row = wn + nt * 16 + l16;
      bh[nt] = frag_b(&smem[buf][2][row * GPITCH], sel);
      bl[nt] = frag_b(&smem[buf][3][row * GPITCH], sel);
    }

    // ---- bf16x3: Y += Ah*Bh + Ah*Bl + Al*Bh ----
#pragma unroll
    for (int mt = 0; mt < 4; ++mt)
#pragma unroll
      for (int nt = 0; nt < 2; ++nt) {
        acc[mt][nt] = __builtin_amdgcn_wmma_f32_16x16x32_bf16(
            false, ah[mt], false, bh[nt], (short)0, acc[mt][nt], false, false);
        acc[mt][nt] = __builtin_amdgcn_wmma_f32_16x16x32_bf16(
            false, ah[mt], false, bl[nt], (short)0, acc[mt][nt], false, false);
        acc[mt][nt] = __builtin_amdgcn_wmma_f32_16x16x32_bf16(
            false, al[mt], false, bh[nt], (short)0, acc[mt][nt], false, false);
      }
    __syncthreads();  // protects smem[buf] before it is refilled next+1 iter
  }

  // ---- store: VGPR g holds row (g + 8*sel), col = l16 within 16x16 tile ----
#pragma unroll
  for (int mt = 0; mt < 4; ++mt)
#pragma unroll
    for (int nt = 0; nt < 2; ++nt) {
      const int col = bn + wn + nt * 16 + l16;
#pragma unroll
      for (int g = 0; g < 8; ++g) {
        const int row = bm + wm + mt * 16 + sel * 8 + g;
        Y[(size_t)row * 4096 + col] = acc[mt][nt][g];
      }
    }
}

// ---------------------------------------------------------------------------
// Host orchestration (graph-capture safe: kernels + D2D memcpyAsync only).
// ---------------------------------------------------------------------------
extern "C" void kernel_launch(void* const* d_in, const int* in_sizes, int n_in,
                              void* d_out, int out_size, void* d_ws, size_t ws_size,
                              hipStream_t stream) {
  (void)in_sizes; (void)n_in; (void)out_size; (void)ws_size;
  const float* x = (const float*)d_in[0];
  const float* cores[12];
  for (int i = 0; i < 12; ++i) cores[i] = (const float*)d_in[1 + i];
  float* Y = (float*)d_out;

  char* ws = (char*)d_ws;
  float*  Pa  = (float*)(ws);                      // 256 KB (64*64*16 f32)
  float*  Pb  = (float*)(ws + (256 << 10));        // 256 KB
  float*  Sa  = (float*)(ws + (512 << 10));        // 64 KB (16*16*64 f32)
  float*  Sb  = (float*)(ws + (576 << 10));        // 64 KB
  bf16_t* Xh  = (bf16_t*)(ws + (640 << 10));       // 8 MB each plane
  bf16_t* Xl  = Xh  + (size_t)4096 * 1024;
  bf16_t* Wth = Xl  + (size_t)4096 * 1024;
  bf16_t* Wtl = Wth + (size_t)4096 * 1024;

  // Prefix chain: core0 flat layout == P1 layout (copy), then cores 1..5.
  (void)hipMemcpyAsync(Pa, cores[0], 64 * sizeof(float), hipMemcpyDeviceToDevice, stream);
  float* pin = Pa; float* pout = Pb;
  int IJ = 2;
  for (int k = 1; k <= 5; ++k) {
    const int total = 4 * IJ * IJ * 16;
    const int grid = (total + 255) / 256;
    prefix_step_kernel<<<grid, 256, 0, stream>>>(pout, pin, cores[k], IJ);
    float* t = pin; pin = pout; pout = t;
    IJ *= 2;
  }

  // Suffix chain: core11 flat layout == S init layout (copy), then 10..6.
  (void)hipMemcpyAsync(Sa, cores[11], 32 * sizeof(float), hipMemcpyDeviceToDevice, stream);
  float* sin_ = Sa; float* sout = Sb;
  int Iin = 1, Jin = 2;
  for (int s = 0; s < 5; ++s) {
    const int ci = 10 - s;
    const int m = (ci == 10) ? 1 : 2;
    const int total = 16 * (m * Iin) * (2 * Jin);
    const int grid = (total + 255) / 256;
    suffix_step_kernel<<<grid, 256, 0, stream>>>(sout, sin_, cores[ci], m, Iin, Jin);
    float* t = sin_; sin_ = sout; sout = t;
    Iin *= m; Jin *= 2;
  }

  build_w_kernel<<<64, 256, 0, stream>>>(pin, sin_, Wth, Wtl);
  split_x_kernel<<<16384, 256, 0, stream>>>(x, Xh, Xl);
  gemm_kernel<<<dim3(32, 32), 256, 0, stream>>>(Xh, Xl, Wth, Wtl, Y);
}
